// ScalarDistanceDeepSet_66271345377322
// MI455X (gfx1250) — compile-verified, matrix operand order checked by reading, not run on hardware
//
#include <hip/hip_runtime.h>

typedef __attribute__((ext_vector_type(16))) _Float16 v16h;
typedef __attribute__((ext_vector_type(8)))  float    v8f;

#define NDIM   256
#define BDIM   32
#define NPAIRS 32640          // 256*255/2
#define PHI1   64
#define PHI2   128
#define RHO1   256
#define RHO2   128
#define OUTD   64
#define TILES_PER_WAVE 15
#define WAVES_PER_BLOCK 8
#define TILES_PER_BLOCK (TILES_PER_WAVE*WAVES_PER_BLOCK)   // 120
#define CHUNKS 17                                          // 17*120 = 2040 tiles = 32640 pairs

// ---------------- kernel 0: pair index table + zero pooled accumulator ----
__global__ __launch_bounds__(256) void k0_init(int* __restrict__ pairIdx,
                                               float* __restrict__ pooled) {
  int tid = blockIdx.x * 256 + threadIdx.x;       // 65536 threads over (i,j)
  int i = tid >> 8, j = tid & 255;
  if (j > i) {
    int p = i * NDIM - (i * (i + 1)) / 2 + (j - i - 1);   // triu_indices order
    pairIdx[p] = (i << 8) | j;
  }
  if (tid < BDIM * PHI2) pooled[tid] = 0.f;
}

// ---------------- kernel 1: per-pair phi MLP via WMMA + masked pooling ----
// Register budget: bf 128 + a2 16 + acc 32 + pacc 8 + temps -> < 256 VGPRs
// (avoids s_set_vgpr_msb churn of the >256-VGPR variant, better occupancy).
__global__ __launch_bounds__(256) void k1_phi(
    const float* __restrict__ dm, const int* __restrict__ lengths,
    const float* __restrict__ W1, const float* __restrict__ b1,
    const float* __restrict__ W2, const float* __restrict__ b2,
    const int* __restrict__ pairIdx, float* __restrict__ pooled) {
  __shared__ _Float16 sW2[PHI1 * PHI2];   // 16 KB
  __shared__ float sW1[PHI1], sb1[PHI1], sb2[PHI2];

  const int tid = threadIdx.x;
  for (int idx = tid; idx < PHI1 * PHI2; idx += 256) sW2[idx] = (_Float16)W2[idx];
  if (tid < PHI1) { sW1[tid] = W1[tid]; sb1[tid] = b1[tid]; }
  if (tid < PHI2) sb2[tid] = b2[tid];
  __syncthreads();

  const int b    = blockIdx.x;
  const int wave = tid >> 5;
  const int lane = tid & 31;
  const int mrow = lane & 15;            // row (for A) / col-in-tile (for B,D)
  const int kb   = (lane >> 4) * 8;      // K sub-group base per ISA 16-bit layout
  const int hi8  = (lane >> 4) * 8;      // D rows M = r + hi8
  const int len  = lengths[b];
  const float* dmb = dm + ((size_t)b << 16);

  // Register-resident B fragments of W2 (64x128): 2 k-steps x 8 n-tiles,
  // reused across all 15 tiles of this wave (zero LDS traffic in hot loop).
  // B layout mirrors A: lane holds column (mrow), halves pack K groups.
  v16h bf[2][8];
  for (int ks = 0; ks < 2; ++ks) {
    for (int nt = 0; nt < 8; ++nt) {
      const int col = nt * 16 + mrow;
      v16h t;
#pragma unroll
      for (int jj = 0; jj < 16; ++jj) {
        const int K = ks * 32 + kb + (jj < 8 ? jj : 8 + jj);
        t[jj] = sW2[K * PHI2 + col];
      }
      bf[ks][nt] = t;
    }
  }

  float pacc[8];
#pragma unroll
  for (int nt = 0; nt < 8; ++nt) pacc[nt] = 0.f;

  for (int it = 0; it < TILES_PER_WAVE; ++it) {
    const int tile = blockIdx.y * TILES_PER_BLOCK + wave * TILES_PER_WAVE + it;
    const int p  = tile * 16 + mrow;
    const int pc = pairIdx[p];
    const int ii = pc >> 8, jc = pc & 255;
    const float s = dmb[(ii << 8) | jc];
    const int msk = (jc < len) ? 1 : 0;              // i < j < len implied
    const unsigned mb = (unsigned)__ballot(msk) & 0xFFFFu;  // row mask bits

    // Build A fragments for both k-steps once per tile (16 VGPRs):
    // h[m][K] = relu(s*W1[K] + b1[K]) per ISA 16-bit A layout.
    v16h a2[2];
#pragma unroll
    for (int ks = 0; ks < 2; ++ks) {
      v16h a;
#pragma unroll
      for (int jj = 0; jj < 16; ++jj) {
        const int K = ks * 32 + kb + (jj < 8 ? jj : 8 + jj);
        a[jj] = (_Float16)fmaxf(s * sW1[K] + sb1[K], 0.f);
      }
      a2[ks] = a;
    }

    // Two passes of 4 n-tiles -> only 32 accumulator VGPRs live at a time.
#pragma unroll
    for (int pass = 0; pass < 2; ++pass) {
      v8f acc[4];
#pragma unroll
      for (int q = 0; q < 4; ++q) acc[q] = (v8f){0, 0, 0, 0, 0, 0, 0, 0};

#pragma unroll
      for (int ks = 0; ks < 2; ++ks) {
#pragma unroll
        for (int q = 0; q < 4; ++q) {
          acc[q] = __builtin_amdgcn_wmma_f32_16x16x32_f16(
              false, a2[ks], false, bf[ks][pass * 4 + q], (short)0, acc[q],
              false, false);
        }
      }

      // Masked ReLU(+b2) column reduction:
      // D VGPR r -> row M = r + hi8, col N = mrow of n-tile (pass*4+q).
#pragma unroll
      for (int q = 0; q < 4; ++q) {
        const int nt = pass * 4 + q;
        const float bias = sb2[nt * 16 + mrow];
        float sum = 0.f;
#pragma unroll
        for (int r = 0; r < 8; ++r) {
          if ((mb >> (r + hi8)) & 1u) sum += fmaxf(acc[q][r] + bias, 0.f);
        }
        pacc[nt] += sum;
      }
    }
  }

  // Combine the two half-lane row groups, then accumulate globally.
#pragma unroll
  for (int nt = 0; nt < 8; ++nt) {
    float v = pacc[nt];
    v += __shfl_xor(v, 16, 32);
    if (lane < 16) atomicAdd(&pooled[b * PHI2 + nt * 16 + mrow], v);
  }
}

// ---------------- kernel 2: rho MLP (tiny, single block, LDS staged) ------
__global__ __launch_bounds__(256) void k2_rho(
    const float* __restrict__ pooled,
    const float* __restrict__ W3, const float* __restrict__ b3,
    const float* __restrict__ W4, const float* __restrict__ b4,
    const float* __restrict__ W5, const float* __restrict__ b5,
    float* __restrict__ out) {
  __shared__ float sA[BDIM * PHI2];   // pooled, later reused for r2 (16 KB)
  __shared__ float sR1[BDIM * RHO1];  // 32 KB
  const int tid = threadIdx.x;

  for (int idx = tid; idx < BDIM * PHI2; idx += 256) sA[idx] = pooled[idx];
  __syncthreads();

  // r1 = relu(pooled @ W3 + b3)   (32 x 256)
  for (int idx = tid; idx < BDIM * RHO1; idx += 256) {
    const int bb = idx >> 8, n = idx & 255;
    float acc = b3[n];
    for (int k = 0; k < PHI2; ++k) acc += sA[bb * PHI2 + k] * W3[k * RHO1 + n];
    sR1[idx] = fmaxf(acc, 0.f);
  }
  __syncthreads();

  // r2 = relu(r1 @ W4 + b4)       (32 x 128) -> overwrite sA
  for (int idx = tid; idx < BDIM * RHO2; idx += 256) {
    const int bb = idx >> 7, n = idx & 127;
    float acc = b4[n];
    for (int k = 0; k < RHO1; ++k) acc += sR1[bb * RHO1 + k] * W4[k * RHO2 + n];
    sA[idx] = fmaxf(acc, 0.f);
  }
  __syncthreads();

  // out = r2 @ W5 + b5            (32 x 64)
  for (int idx = tid; idx < BDIM * OUTD; idx += 256) {
    const int bb = idx >> 6, n = idx & 63;
    float acc = b5[n];
    for (int k = 0; k < RHO2; ++k) acc += sA[bb * RHO2 + k] * W5[k * OUTD + n];
    out[idx] = acc;
  }
}

extern "C" void kernel_launch(void* const* d_in, const int* in_sizes, int n_in,
                              void* d_out, int out_size, void* d_ws, size_t ws_size,
                              hipStream_t stream) {
  const float* dm      = (const float*)d_in[0];
  const int*   lengths = (const int*)d_in[1];
  const float* W1 = (const float*)d_in[2];
  const float* b1 = (const float*)d_in[3];
  const float* W2 = (const float*)d_in[4];
  const float* b2 = (const float*)d_in[5];
  const float* W3 = (const float*)d_in[6];
  const float* b3 = (const float*)d_in[7];
  const float* W4 = (const float*)d_in[8];
  const float* b4 = (const float*)d_in[9];
  const float* W5 = (const float*)d_in[10];
  const float* b5 = (const float*)d_in[11];
  float* out = (float*)d_out;

  char* ws = (char*)d_ws;
  float* pooled = (float*)ws;                           // 32*128 f32 = 16 KB
  int* pairIdx  = (int*)(ws + 16384);                   // 32640 ints

  k0_init<<<256, 256, 0, stream>>>(pairIdx, pooled);
  k1_phi<<<dim3(BDIM, CHUNKS), 256, 0, stream>>>(dm, lengths, W1, b1, W2, b2,
                                                 pairIdx, pooled);
  k2_rho<<<1, 256, 0, stream>>>(pooled, W3, b3, W4, b4, W5, b5, out);
}
